// HCMamba_CSI_HPE_28819230556871
// MI455X (gfx1250) — compile-verified, matrix operand order checked by reading, not run on hardware
//
#include <hip/hip_runtime.h>
#include <math.h>

typedef __attribute__((ext_vector_type(16))) _Float16 v16h;
typedef __attribute__((ext_vector_type(8)))  _Float16 v8h;
typedef __attribute__((ext_vector_type(8)))  float    v8f;

#define kD      512
#define kDin    1024
#define kState  16
#define kDtRank 32
#define kBatch  8
#define kL      512
#define kLlt    128
#define kFdim   342
#define kFpad   352   /* 342 padded to multiple of 32 */
#define LN_EPS  1e-5f
#define RMS_EPS 1.1920929e-7f

__device__ __forceinline__ float hc_sigm(float x) { return 1.0f / (1.0f + __expf(-x)); }

__device__ __forceinline__ v16h hc_pack16f(float4 a0, float4 a1, float4 a2, float4 a3) {
  v16h r;
  r[0]  = (_Float16)a0.x; r[1]  = (_Float16)a0.y; r[2]  = (_Float16)a0.z; r[3]  = (_Float16)a0.w;
  r[4]  = (_Float16)a1.x; r[5]  = (_Float16)a1.y; r[6]  = (_Float16)a1.z; r[7]  = (_Float16)a1.w;
  r[8]  = (_Float16)a2.x; r[9]  = (_Float16)a2.y; r[10] = (_Float16)a2.z; r[11] = (_Float16)a2.w;
  r[12] = (_Float16)a3.x; r[13] = (_Float16)a3.y; r[14] = (_Float16)a3.z; r[15] = (_Float16)a3.w;
  return r;
}

__device__ __forceinline__ v16h hc_pack16h(v8h lo, v8h hi) {
  v16h r;
#pragma unroll
  for (int j = 0; j < 8; ++j) { r[j] = lo[j]; r[8 + j] = hi[j]; }
  return r;
}

// ---------------------------------------------------------------------------
// WMMA GEMM:  C[M,N] = act( A[M,K]_f32 @ W[N,K]_f16^T + bias ) + resid
// One wave computes a 16(M) x 32(N) tile; A fragment reused across two WMMAs.
// REQUIRES: K % 32 == 0, lda % 8 == 0, ldw % 8 == 0, 16B-aligned bases.
// ACT: 0=none, 1=softplus, 2=gelu(exact)
// ---------------------------------------------------------------------------
template <int ACT>
__global__ __launch_bounds__(128) void hc_wmma_gemm_k(
    const float* __restrict__ A, int lda, const _Float16* __restrict__ W, int ldw,
    float* __restrict__ C, int ldc, const float* __restrict__ bias,
    const float* __restrict__ resid, int ldres, int M, int N, int K) {
  const int lane = threadIdx.x & 31;
  const int wave = threadIdx.x >> 5;
  const int mt = blockIdx.y * 4 + wave;
  if (mt * 16 >= M) return;

  const int l15 = lane & 15;
  const int khalf = (lane & 16) ? 8 : 0;  // ISA 16-bit fragment K split
  const int mrow = mt * 16 + l15;
  const int n0 = blockIdx.x * 32 + l15;
  const int n1 = n0 + 16;

  // clamp OOB rows/cols to a valid row; results masked at store
  const float*    __restrict__ Ar = A + (size_t)(mrow < M ? mrow : M - 1) * (size_t)lda;
  const _Float16* __restrict__ W0 = W + (size_t)(n0 < N ? n0 : N - 1) * (size_t)ldw;
  const _Float16* __restrict__ W1 = W + (size_t)(n1 < N ? n1 : N - 1) * (size_t)ldw;

  v8f acc0 = {}, acc1 = {};
  for (int kb = 0; kb < K; kb += 32) {
    __builtin_prefetch(Ar + kb + 128, 0, 1);  // global_prefetch_b8
    const float4* ap = (const float4*)(Ar + kb + khalf);
    const float4* aq = (const float4*)(Ar + kb + 16 + khalf);
    const v8h*    bp = (const v8h*)(W0 + kb + khalf);
    const v8h*    bq = (const v8h*)(W0 + kb + 16 + khalf);
    const v8h*    cp = (const v8h*)(W1 + kb + khalf);
    const v8h*    cq = (const v8h*)(W1 + kb + 16 + khalf);
    v16h af = hc_pack16f(ap[0], ap[1], aq[0], aq[1]);
    v16h bf = hc_pack16h(bp[0], bq[0]);
    v16h cf = hc_pack16h(cp[0], cq[0]);
    acc0 = __builtin_amdgcn_wmma_f32_16x16x32_f16(false, af, false, bf,
                                                  (short)0, acc0, false, false);
    acc1 = __builtin_amdgcn_wmma_f32_16x16x32_f16(false, af, false, cf,
                                                  (short)0, acc1, false, false);
  }

  const int mb = mt * 16 + ((lane & 16) ? 8 : 0);
#pragma unroll
  for (int r = 0; r < 8; ++r) {
    const int m = mb + r;
    if (m >= M) continue;
#pragma unroll
    for (int half = 0; half < 2; ++half) {
      const int n = (half ? n1 : n0);
      if (n < N) {
        float v = half ? acc1[r] : acc0[r];
        if (bias) v += bias[n];
        if (ACT == 1) v = (v > 20.0f) ? v : log1pf(__expf(v));
        else if (ACT == 2) v = 0.5f * v * (1.0f + erff(v * 0.70710678118654752f));
        if (resid) v += resid[(size_t)m * ldres + n];
        C[(size_t)m * ldc + n] = v;
      }
    }
  }
}

// generic f32 -> f16 weight conversion
__global__ void hc_f32_to_f16_k(const float* __restrict__ x, _Float16* __restrict__ o, int n) {
  int tid = blockIdx.x * blockDim.x + threadIdx.x;
  if (tid < n) o[tid] = (_Float16)x[tid];
}

// proj_in_w (512 x 342) -> zero-padded f16 (512 x kFpad)
__global__ void hc_pad_w16_k(const float* __restrict__ w, _Float16* __restrict__ wp) {
  int tid = blockIdx.x * blockDim.x + threadIdx.x;
  if (tid >= kD * kFpad) return;
  int col = tid % kFpad, row = tid / kFpad;
  wp[tid] = (col < kFdim) ? (_Float16)w[(size_t)row * kFdim + col] : (_Float16)0.0f;
}

// (b,tx,f,l) -> (b*l, kFpad) zero-padded row-major activations for proj_in
__global__ void hc_repack_x_k(const float* __restrict__ x, float* __restrict__ xf) {
  int tid = blockIdx.x * blockDim.x + threadIdx.x;
  const int total = kBatch * kL * kFpad;
  if (tid >= total) return;
  int col = tid % kFpad;
  int row = tid / kFpad;
  if (col >= kFdim) { xf[tid] = 0.0f; return; }
  int l = row % kL, b = row / kL;
  int tx = col / 114, f = col % 114;
  xf[tid] = x[(((size_t)b * 3 + tx) * 114 + f) * kL + l];
}

__global__ void hc_glu_k(const float* __restrict__ up, float* __restrict__ o, int rows) {
  int tid = blockIdx.x * blockDim.x + threadIdx.x;
  if (tid >= rows * kDin) return;
  int i = tid % kDin, r = tid / kDin;
  float v = up[(size_t)r * (2 * kDin) + i];
  float g = up[(size_t)r * (2 * kDin) + kDin + i];
  o[tid] = v * hc_sigm(g);
}

__global__ __launch_bounds__(256) void hc_layernorm_k(
    const float* __restrict__ x, const float* __restrict__ w,
    const float* __restrict__ b, float* __restrict__ o, int dim, float eps) {
  __shared__ float red[256];
  const size_t row = blockIdx.x;
  const float* xr = x + row * dim;
  float* orow = o + row * dim;
  float s = 0.0f;
  for (int i = threadIdx.x; i < dim; i += 256) s += xr[i];
  red[threadIdx.x] = s; __syncthreads();
  for (int st = 128; st > 0; st >>= 1) {
    if (threadIdx.x < st) red[threadIdx.x] += red[threadIdx.x + st];
    __syncthreads();
  }
  const float mean = red[0] / dim;
  __syncthreads();
  float v = 0.0f;
  for (int i = threadIdx.x; i < dim; i += 256) { float d = xr[i] - mean; v += d * d; }
  red[threadIdx.x] = v; __syncthreads();
  for (int st = 128; st > 0; st >>= 1) {
    if (threadIdx.x < st) red[threadIdx.x] += red[threadIdx.x + st];
    __syncthreads();
  }
  const float inv = rsqrtf(red[0] / dim + eps);
  for (int i = threadIdx.x; i < dim; i += 256)
    orow[i] = (xr[i] - mean) * inv * w[i] + b[i];
}

__global__ __launch_bounds__(256) void hc_rmsnorm_k(
    const float* __restrict__ x, const float* __restrict__ w,
    float* __restrict__ o, int dim) {
  __shared__ float red[256];
  const size_t row = blockIdx.x;
  const float* xr = x + row * dim;
  float* orow = o + row * dim;
  float s = 0.0f;
  for (int i = threadIdx.x; i < dim; i += 256) { float v = xr[i]; s += v * v; }
  red[threadIdx.x] = s; __syncthreads();
  for (int st = 128; st > 0; st >>= 1) {
    if (threadIdx.x < st) red[threadIdx.x] += red[threadIdx.x + st];
    __syncthreads();
  }
  const float inv = rsqrtf(red[0] / dim + RMS_EPS);
  for (int i = threadIdx.x; i < dim; i += 256) orow[i] = xr[i] * inv * w[i];
}

// causal depthwise conv k=4 (+bias) + SiLU on xm = xr[:, :kDin]
__global__ void hc_dwconv_silu_k(const float* __restrict__ xr, const float* __restrict__ w,
                                 const float* __restrict__ b, float* __restrict__ xc,
                                 int Bb, int Ll) {
  int tid = blockIdx.x * blockDim.x + threadIdx.x;
  if (tid >= Bb * Ll * kDin) return;
  int d = tid % kDin;
  int row = tid / kDin;
  int l = row % Ll, bb = row / Ll;
  float acc = b[d];
#pragma unroll
  for (int j = 0; j < 4; ++j) {
    int ls = l - 3 + j;
    if (ls >= 0) acc += w[d * 4 + j] * xr[((size_t)bb * Ll + ls) * (2 * kDin) + d];
  }
  xc[(size_t)row * kDin + d] = acc * hc_sigm(acc);
}

// selective scan, state n=16 in registers; fused * silu(res) gate
__global__ void hc_scan_gate_k(const float* __restrict__ xc, const float* __restrict__ dtb,
                               const float* __restrict__ dbl, const float* __restrict__ xr,
                               const float* __restrict__ A_log, const float* __restrict__ Dv,
                               float* __restrict__ y, int Bb, int Ll) {
  int tid = blockIdx.x * blockDim.x + threadIdx.x;
  if (tid >= Bb * kDin) return;
  int d = tid % kDin, b = tid / kDin;
  float a[kState], h[kState];
#pragma unroll
  for (int n = 0; n < kState; ++n) { a[n] = -__expf(A_log[d * kState + n]); h[n] = 0.0f; }
  const float Dd = Dv[d];
  for (int l = 0; l < Ll; ++l) {
    const size_t row = (size_t)b * Ll + l;
    const float dt = dtb[row * kDin + d];
    const float u = xc[row * kDin + d];
    const float du = dt * u;
    const float* bc = dbl + row * 64;
    float yv = 0.0f;
#pragma unroll
    for (int n = 0; n < kState; ++n) {
      h[n] = __expf(dt * a[n]) * h[n] + du * bc[32 + n];
      yv += h[n] * bc[48 + n];
    }
    const float r = xr[row * (2 * kDin) + kDin + d];
    y[row * kDin + d] = (yv + u * Dd) * (r * hc_sigm(r));
  }
}

// depthwise strided downsample conv (k=stride=4)
__global__ void hc_dsconv_k(const float* __restrict__ s, const float* __restrict__ w,
                            const float* __restrict__ b, float* __restrict__ lt) {
  int tid = blockIdx.x * blockDim.x + threadIdx.x;
  if (tid >= kBatch * kLlt * kD) return;
  int d = tid % kD;
  int lo = (tid / kD) % kLlt;
  int bb = tid / (kD * kLlt);
  float acc = b[d];
#pragma unroll
  for (int j = 0; j < 4; ++j)
    acc += w[d * 4 + j] * s[((size_t)bb * kL + lo * 4 + j) * kD + d];
  lt[tid] = acc;
}

// upsample branch evaluated only at the final token:
// interp positions 510/511 both collapse exactly to lt[:,127,:] (lo=hi=127),
// and the j=2 conv tap reads the zero pad.
__global__ void hc_upx_last_k(const float* __restrict__ lt, const float* __restrict__ upw,
                              const float* __restrict__ upb, float* __restrict__ o) {
  int tid = blockIdx.x * blockDim.x + threadIdx.x;
  if (tid >= kBatch * kD) return;
  int co = tid % kD, b = tid / kD;
  const float* row = lt + ((size_t)b * kLlt + (kLlt - 1)) * kD;
  float acc = upb[co];
  for (int ci = 0; ci < kD; ++ci) {
    const float* wp = upw + ((size_t)co * kD + ci) * 3;
    acc += (wp[0] + wp[1]) * row[ci];
  }
  o[tid] = acc;
}

__global__ void hc_fuse_last_k(const float* __restrict__ s, const float* __restrict__ upxl,
                               const float* __restrict__ res, float* __restrict__ o) {
  int tid = blockIdx.x * blockDim.x + threadIdx.x;
  if (tid >= kBatch * kD) return;
  int i = tid % kD, b = tid / kD;
  size_t r = ((size_t)b * kL + (kL - 1)) * kD + i;
  o[tid] = s[r] + upxl[tid] + res[r];
}

static void hc_gemm(int act, const float* A, int lda, const _Float16* W, int ldw,
                    float* C, int ldc, const float* bias, const float* resid, int ldres,
                    int M, int N, int K, hipStream_t stream) {
  dim3 grid((N + 31) / 32, (M + 63) / 64);
  dim3 blk(128);
  switch (act) {
    case 1:
      hc_wmma_gemm_k<1><<<grid, blk, 0, stream>>>(A, lda, W, ldw, C, ldc, bias, resid, ldres, M, N, K);
      break;
    case 2:
      hc_wmma_gemm_k<2><<<grid, blk, 0, stream>>>(A, lda, W, ldw, C, ldc, bias, resid, ldres, M, N, K);
      break;
    default:
      hc_wmma_gemm_k<0><<<grid, blk, 0, stream>>>(A, lda, W, ldw, C, ldc, bias, resid, ldres, M, N, K);
      break;
  }
}

extern "C" void kernel_launch(void* const* d_in, const int* in_sizes, int n_in,
                              void* d_out, int out_size, void* d_ws, size_t ws_size,
                              hipStream_t stream) {
  (void)in_sizes; (void)n_in; (void)out_size; (void)ws_size;
  auto F = [&](int i) { return (const float*)d_in[i]; };

  // ---- workspace carve-up: f32 arena, then f16 weight arena ----
  float* ws = (float*)d_ws;
  size_t off = 0;
  auto alloc = [&](size_t n) { float* p = ws + off; off += (n + 63) & ~(size_t)63; return p; };
  float* xf   = alloc((size_t)kBatch * kL * kFpad);        // 4096 x 352 (padded)
  float* e    = alloc((size_t)kBatch * kL * kD);           // residual / GSFM out
  float* t    = alloc((size_t)kBatch * kL * kD);           // norm temp
  float* s    = alloc((size_t)kBatch * kL * kD);           // STM stream
  float* xr   = alloc((size_t)kBatch * kL * 2 * kDin);     // in_proj out / GSFM up
  float* xc   = alloc((size_t)kBatch * kL * kDin);         // conv+silu / GLU out
  float* dbl  = alloc((size_t)kBatch * kL * 64);           // x_proj out (dt|B|C)
  float* dtb  = alloc((size_t)kBatch * kL * kDin);         // softplus dt
  float* ybuf = alloc((size_t)kBatch * kL * kDin);         // scan output (gated)
  float* lt   = alloc((size_t)kBatch * kLlt * kD);         // downsampled stream
  float* upxl = alloc((size_t)kBatch * kD);
  float* last = alloc((size_t)kBatch * kD);
  float* lastn= alloc((size_t)kBatch * kD);
  float* h1   = alloc((size_t)kBatch * (kD / 2));

  _Float16* harena = (_Float16*)(ws + off);
  size_t hoff = 0;
  auto halloc = [&](size_t n) { _Float16* p = harena + hoff; hoff += (n + 127) & ~(size_t)127; return p; };

  const int rowsL = kBatch * kL;  // 4096
  const int EW = 256;

  // one-pass f32 -> f16 conversion of every GEMM weight (amortized over 256 M-tiles)
  auto cvtw = [&](const float* w, size_t n) {
    _Float16* p = halloc(n);
    hc_f32_to_f16_k<<<((int)n + EW - 1) / EW, EW, 0, stream>>>(w, p, (int)n);
    return p;
  };

  _Float16* w1p = halloc((size_t)kD * kFpad);  // padded proj_in_w (f16)
  hc_pad_w16_k<<<(kD * kFpad + EW - 1) / EW, EW, 0, stream>>>(F(1), w1p);
  _Float16* wup   = cvtw(F(3), (size_t)4 * kD * kD);        // gsfm_up   2048x512
  _Float16* wdown = cvtw(F(5), (size_t)kD * 2 * kD);        // gsfm_down 512x1024
  const int bases[6] = {9, 19, 29, 39, 49, 59};
  _Float16 *w_in[6], *w_x[6], *w_dt[6], *w_out[6];
  for (int i = 0; i < 6; ++i) {
    w_in[i]  = cvtw(F(bases[i] + 0), (size_t)2 * kDin * kD);       // 2048x512
    w_x[i]   = cvtw(F(bases[i] + 3), (size_t)64 * kDin);           // 64x1024
    w_dt[i]  = cvtw(F(bases[i] + 4), (size_t)kDin * kDtRank);      // 1024x32
    w_out[i] = cvtw(F(bases[i] + 8), (size_t)kD * kDin);           // 512x1024
  }
  _Float16* wfc1 = cvtw(F(75), (size_t)(kD / 2) * kD);      // 256x512
  _Float16* wfc2 = cvtw(F(77), (size_t)102 * (kD / 2));     // 102x256

  // ---- mamba block runner ----
  auto run_mamba = [&](const float* xin, float* xout, int Bb, int Ll, int i) {
    const int pb = bases[i];
    const int rows = Bb * Ll;
    hc_rmsnorm_k<<<rows, EW, 0, stream>>>(xin, F(pb + 9), t, kD);
    hc_gemm(0, t, kD, w_in[i], kD, xr, 2 * kDin, nullptr, nullptr, 0, rows, 2 * kDin, kD, stream);
    hc_dwconv_silu_k<<<(rows * kDin + EW - 1) / EW, EW, 0, stream>>>(xr, F(pb + 1), F(pb + 2), xc, Bb, Ll);
    hc_gemm(0, xc, kDin, w_x[i], kDin, dbl, 64, nullptr, nullptr, 0, rows, 64, kDin, stream);
    hc_gemm(1, dbl, 64, w_dt[i], kDtRank, dtb, kDin, F(pb + 5), nullptr, 0, rows, kDin, kDtRank, stream);
    hc_scan_gate_k<<<(Bb * kDin + EW - 1) / EW, EW, 0, stream>>>(xc, dtb, dbl, xr, F(pb + 6), F(pb + 7), ybuf, Bb, Ll);
    hc_gemm(0, ybuf, kDin, w_out[i], kDin, xout, kD, nullptr, xin, kD, rows, kD, kDin, stream);
  };

  // ---- stage 1: embed + GSFM ----
  hc_repack_x_k<<<(rowsL * kFpad + EW - 1) / EW, EW, 0, stream>>>(F(0), xf);
  hc_gemm(0, xf, kFpad, w1p, kFpad, e, kD, F(2), nullptr, 0, rowsL, kD, kFpad, stream);
  hc_gemm(0, e, kD, wup, kD, xr, 2 * kDin, F(4), nullptr, 0, rowsL, 2 * kDin, kD, stream);
  hc_glu_k<<<(rowsL * kDin + EW - 1) / EW, EW, 0, stream>>>(xr, xc, rowsL);
  hc_gemm(0, xc, kDin, wdown, kDin, t, kD, F(6), e, kD, rowsL, kD, kDin, stream);
  hc_layernorm_k<<<rowsL, EW, 0, stream>>>(t, F(7), F(8), e, kD, LN_EPS);

  // ---- stage 2: 4 STM mamba blocks (e preserved as residual for the head) ----
  run_mamba(e, s, kBatch, kL, 0);
  run_mamba(s, s, kBatch, kL, 1);
  run_mamba(s, s, kBatch, kL, 2);
  run_mamba(s, s, kBatch, kL, 3);

  // ---- stage 3: downsample + 2 LTM mamba blocks ----
  hc_dsconv_k<<<(kBatch * kLlt * kD + EW - 1) / EW, EW, 0, stream>>>(s, F(69), F(70), lt);
  run_mamba(lt, lt, kBatch, kLlt, 4);
  run_mamba(lt, lt, kBatch, kLlt, 5);

  // ---- stage 4: upsample branch at last token only + head ----
  hc_upx_last_k<<<(kBatch * kD + EW - 1) / EW, EW, 0, stream>>>(lt, F(71), F(72), upxl);
  hc_fuse_last_k<<<(kBatch * kD + EW - 1) / EW, EW, 0, stream>>>(s, upxl, e, last);
  hc_layernorm_k<<<kBatch, EW, 0, stream>>>(last, F(73), F(74), lastn, kD, LN_EPS);
  hc_gemm(2, lastn, kD, wfc1, kD, h1, kD / 2, F(76), nullptr, 0, kBatch, kD / 2, kD, stream);
  hc_gemm(0, h1, kD / 2, wfc2, kD / 2, (float*)d_out, 102, F(78), nullptr, 0, kBatch, 102, kD / 2, stream);
}
